// SAGPool_44178033606815
// MI455X (gfx1250) — compile-verified, matrix operand order checked by reading, not run on hardware
//
#include <hip/hip_runtime.h>
#include <hip/hip_bf16.h>
#include <math.h>

// ---------------------------------------------------------------------------
// SAGPool GNN forward for MI455X (gfx1250, wave32).
//   - edge scatter via L2-resident fp32 atomics (x table fits in 192MB L2)
//   - SAGEConv linear layers via v_wmma_f32_16x16x32_f16 (fp32 accumulate)
//   - exact top_k tie semantics (rank by score desc, index asc)
// ---------------------------------------------------------------------------

typedef _Float16 half_t;
typedef __attribute__((ext_vector_type(16))) _Float16 v16h;
typedef __attribute__((ext_vector_type(8)))  _Float16 v8h;
typedef __attribute__((ext_vector_type(8)))  float    v8f;

#define BGRAPHS 200
#define NEDGES  1600000
#define HDIM    128
#define NMAX    100000
#define BN_EPS  1e-5f

// ------------------------------- utility -----------------------------------

__global__ void zero_f32_kernel(float* __restrict__ p, int n) {
  for (int i = blockIdx.x * blockDim.x + threadIdx.x; i < n;
       i += gridDim.x * blockDim.x)
    p[i] = 0.0f;
}

__global__ void edge_init_kernel(const int* __restrict__ ei,
                                 int* __restrict__ s, int* __restrict__ d,
                                 int ne) {
  int e = blockIdx.x * blockDim.x + threadIdx.x;
  if (e >= ne) return;
  s[e] = ei[e];
  d[e] = ei[ne + e];
}

__global__ void cvt_f16_kernel(const float* __restrict__ in,
                               half_t* __restrict__ out, int n) {
  int i = blockIdx.x * blockDim.x + threadIdx.x;
  if (i < n) out[i] = (half_t)in[i];
}

// Transpose 128x128 fp32 weight (fan_in x fan_out) into f16 [out][in] so that
// WMMA B-tiles load as contiguous 16B chunks per lane.
__global__ void weightT_kernel(const float* __restrict__ W,
                               half_t* __restrict__ WT) {
  int i = blockIdx.x * blockDim.x + threadIdx.x;
  if (i >= HDIM * HDIM) return;
  int n = i >> 7, k = i & 127;
  WT[i] = (half_t)W[k * HDIM + n];
}

// ------------------------- edge message passing ----------------------------
// one wave per edge: lane c moves channels [4c, 4c+4)
__global__ __launch_bounds__(256) void edge_msg_kernel(
    const int* __restrict__ src, const int* __restrict__ dst,
    const float* __restrict__ x, float* __restrict__ msg,
    float* __restrict__ cnt, int ne) {
  int gt = blockIdx.x * blockDim.x + threadIdx.x;
  int e = gt >> 5, lane = gt & 31;
  if (e >= ne) return;
  int s = src[e];
  if (s < 0) return;
  int d = dst[e];
  float4 v = ((const float4*)(x + (size_t)s * HDIM))[lane];
  float* o = msg + (size_t)d * HDIM + lane * 4;
  atomicAdd(o + 0, v.x);
  atomicAdd(o + 1, v.y);
  atomicAdd(o + 2, v.z);
  atomicAdd(o + 3, v.w);
  if (lane == 0) atomicAdd(cnt + d, 1.0f);
}

__global__ void mean_fin_kernel(const float* __restrict__ msg,
                                const float* __restrict__ cnt,
                                half_t* __restrict__ mean16, int total) {
  int i = blockIdx.x * blockDim.x + threadIdx.x;
  if (i >= total) return;
  float c = fmaxf(cnt[i >> 7], 1.0f);
  mean16[i] = (half_t)(msg[i] / c);
}

// ------------------------------ WMMA GEMM ----------------------------------
// h = relu(mean @ Wl + bl + x @ Wr) * rs      (rs = 1/sqrt(per))
// grid.x = n/16 row tiles; 8 waves per block = 8 col tiles of 16.
__device__ inline v8f wmma_acc_128(const half_t* __restrict__ arow,
                                   const half_t* __restrict__ brow,
                                   int kb, v8f acc) {
#pragma unroll
  for (int k0 = 0; k0 < HDIM; k0 += 32) {
    v8h al = *(const v8h*)(arow + k0 + kb);
    v8h ah = *(const v8h*)(arow + k0 + 16 + kb);
    v8h bl = *(const v8h*)(brow + k0 + kb);
    v8h bh = *(const v8h*)(brow + k0 + 16 + kb);
    v16h a, b;
#pragma unroll
    for (int i = 0; i < 8; ++i) {
      a[i] = al[i]; a[8 + i] = ah[i];
      b[i] = bl[i]; b[8 + i] = bh[i];
    }
    acc = __builtin_amdgcn_wmma_f32_16x16x32_f16(false, a, false, b,
                                                 (short)0, acc, false, false);
  }
  return acc;
}

__global__ __launch_bounds__(256) void sage_gemm_kernel(
    const half_t* __restrict__ mean16, const half_t* __restrict__ x16,
    const half_t* __restrict__ wlT, const half_t* __restrict__ wrT,
    const float* __restrict__ bias, float rs, float* __restrict__ h) {
  const int lane = threadIdx.x & 31;
  const int wave = threadIdx.x >> 5;       // output column tile 0..7
  const int m15 = lane & 15;
  const int kb = (lane >> 4) * 8;          // 0 or 8 (A/B 16-bit layout)
  const int row0 = blockIdx.x * 16;
  const int m = row0 + m15;
  const int n = wave * 16 + m15;

  v8f acc = {};
  acc = wmma_acc_128(mean16 + (size_t)m * HDIM, wlT + (size_t)n * HDIM, kb, acc);
  acc = wmma_acc_128(x16 + (size_t)m * HDIM,   wrT + (size_t)n * HDIM, kb, acc);

  const int half8 = (lane >> 4) * 8;       // C/D row offset for lanes 16..31
  float bv = bias[n];
#pragma unroll
  for (int r = 0; r < 8; ++r) {
    float v = fmaxf(acc[r] + bv, 0.0f) * rs;
    h[(size_t)(row0 + half8 + r) * HDIM + n] = v;
  }
}

// ------------------------------ BatchNorm ----------------------------------
__global__ __launch_bounds__(128) void bn_stats_kernel(
    const float* __restrict__ h, float* __restrict__ bnsum,
    float* __restrict__ bnss, int nrows) {
  int c = threadIdx.x;
  float s = 0.0f, ss = 0.0f;
  for (int r = blockIdx.x; r < nrows; r += gridDim.x) {
    float v = h[(size_t)r * HDIM + c];
    s += v;
    ss += v * v;
  }
  atomicAdd(bnsum + c, s);
  atomicAdd(bnss + c, ss);
}

__global__ void bn_apply_kernel(float* __restrict__ h,
                                const float* __restrict__ bnsum,
                                const float* __restrict__ bnss,
                                const float* __restrict__ g,
                                const float* __restrict__ b, float inv_n,
                                int total) {
  int i = blockIdx.x * blockDim.x + threadIdx.x;
  if (i >= total) return;
  int c = i & 127;
  float mu = bnsum[c] * inv_n;
  float var = bnss[c] * inv_n - mu * mu;
  h[i] = (h[i] - mu) * rsqrtf(var + BN_EPS) * g[c] + b[c];
}

// ------------------------- SAGPool score (GCNConv) -------------------------
__global__ __launch_bounds__(256) void xw_kernel(const float* __restrict__ h,
                                                 const float* __restrict__ pw,
                                                 float* __restrict__ xw,
                                                 int nrows) {
  int gt = blockIdx.x * blockDim.x + threadIdx.x;
  int row = gt >> 5, lane = gt & 31;
  if (row >= nrows) return;
  float4 hv = ((const float4*)(h + (size_t)row * HDIM))[lane];
  float4 wv = ((const float4*)pw)[lane];
  float p = hv.x * wv.x + hv.y * wv.y + hv.z * wv.z + hv.w * wv.w;
  p += __shfl_xor(p, 16, 32);
  p += __shfl_xor(p, 8, 32);
  p += __shfl_xor(p, 4, 32);
  p += __shfl_xor(p, 2, 32);
  p += __shfl_xor(p, 1, 32);
  if (lane == 0) xw[row] = p;
}

__global__ void score_edge_kernel(const int* __restrict__ src,
                                  const int* __restrict__ dst,
                                  const float* __restrict__ xw,
                                  const float* __restrict__ cnt,
                                  float* __restrict__ agg, int ne) {
  int e = blockIdx.x * blockDim.x + threadIdx.x;
  if (e >= ne) return;
  int s = src[e];
  if (s < 0) return;
  int d = dst[e];
  float ds = cnt[s] + 1.0f, dd = cnt[d] + 1.0f;
  atomicAdd(agg + d, xw[s] * rsqrtf(ds * dd));
}

__global__ void score_fin_kernel(const float* __restrict__ agg,
                                 const float* __restrict__ xw,
                                 const float* __restrict__ cnt,
                                 const float* __restrict__ pb,
                                 float* __restrict__ score, int nrows) {
  int i = blockIdx.x * blockDim.x + threadIdx.x;
  if (i >= nrows) return;
  score[i] = agg[i] + xw[i] / (cnt[i] + 1.0f) + pb[0];
}

// --------------------------- per-graph top-k -------------------------------
// Rank = #{i : s_i > s_j or (s_i == s_j and i < j)}  (matches lax.top_k order)
__global__ __launch_bounds__(256) void topk_kernel(
    const float* __restrict__ score, int* __restrict__ newid,
    int* __restrict__ perm, int P, int K) {
  __shared__ float s[512];
  int g = blockIdx.x;
  for (int i = threadIdx.x; i < P; i += blockDim.x) s[i] = score[g * P + i];
  __syncthreads();
  for (int j = threadIdx.x; j < P; j += blockDim.x) {
    float sj = s[j];
    int r = 0;
    for (int i = 0; i < P; ++i) {
      float si = s[i];
      r += (si > sj) || (si == sj && i < j);
    }
    int old = g * P + j;
    if (r < K) {
      int nid = g * K + r;
      newid[old] = nid;
      perm[nid] = old;
    } else {
      newid[old] = -1;
    }
  }
}

__global__ __launch_bounds__(256) void gather_kernel(
    const float* __restrict__ h, const float* __restrict__ score,
    const int* __restrict__ perm, float* __restrict__ xnew, int nnew) {
  int gt = blockIdx.x * blockDim.x + threadIdx.x;
  int row = gt >> 5, lane = gt & 31;
  if (row >= nnew) return;
  int old = perm[row];
  float t = tanhf(score[old]);
  float4 v = ((const float4*)(h + (size_t)old * HDIM))[lane];
  float4 o = {v.x * t, v.y * t, v.z * t, v.w * t};
  ((float4*)(xnew + (size_t)row * HDIM))[lane] = o;
}

__global__ void relabel_kernel(int* __restrict__ src, int* __restrict__ dst,
                               const int* __restrict__ newid, int ne) {
  int e = blockIdx.x * blockDim.x + threadIdx.x;
  if (e >= ne) return;
  int s = src[e];
  if (s < 0) return;
  int sn = newid[s], dn = newid[dst[e]];
  if (sn < 0 || dn < 0) {
    src[e] = -1;
  } else {
    src[e] = sn;
    dst[e] = dn;
  }
}

// ----------------------- global mean/max pooling ---------------------------
__global__ __launch_bounds__(128) void pool_kernel(const float* __restrict__ x,
                                                   float* __restrict__ xs,
                                                   int P) {
  int g = blockIdx.x, c = threadIdx.x;
  const float* base = x + (size_t)g * P * HDIM + c;
  float sum = 0.0f, mx = -3.4e38f;
  for (int i = 0; i < P; ++i) {
    float v = base[(size_t)i * HDIM];
    sum += v;
    mx = fmaxf(mx, v);
  }
  xs[g * 256 + c] += sum / (float)P;
  xs[g * 256 + 128 + c] += mx;
}

// ------------------------------- final MLP ---------------------------------
__global__ __launch_bounds__(256) void mlp_kernel(
    const float* __restrict__ xs, const float* __restrict__ W1,
    const float* __restrict__ b1, const float* __restrict__ W2,
    const float* __restrict__ b2, const float* __restrict__ W3,
    const float* __restrict__ b3, float* __restrict__ out) {
  __shared__ float s0[256];
  __shared__ float s1[128];
  __shared__ float s2[64];
  int g = blockIdx.x, t = threadIdx.x;
  s0[t] = xs[g * 256 + t];
  __syncthreads();
  if (t < 128) {
    float a = b1[t];
    for (int i = 0; i < 256; ++i) a += s0[i] * W1[i * 128 + t];
    s1[t] = fmaxf(a, 0.0f);
  }
  __syncthreads();
  if (t < 64) {
    float a = b2[t];
    for (int i = 0; i < 128; ++i) a += s1[i] * W2[i * 64 + t];
    s2[t] = fmaxf(a, 0.0f);
  }
  __syncthreads();
  if (t < 6) {
    float a = b3[t];
    for (int i = 0; i < 64; ++i) a += s2[i] * W3[i * 6 + t];
    out[g * 6 + t] = a;
  }
}

// ------------------------------- launcher ----------------------------------

static inline size_t align256(size_t x) { return (x + 255) & ~(size_t)255; }

extern "C" void kernel_launch(void* const* d_in, const int* in_sizes, int n_in,
                              void* d_out, int out_size, void* d_ws,
                              size_t ws_size, hipStream_t stream) {
  (void)in_sizes; (void)n_in; (void)out_size; (void)ws_size;

  // --- inputs (setup_inputs dict insertion order) ---
  const float* d_x = (const float*)d_in[0];
  const int* d_ei = (const int*)d_in[1];
  int pi = 2;
  const float *Wl[3], *bl[3], *Wr[3];
  for (int i = 0; i < 3; ++i) {
    Wl[i] = (const float*)d_in[pi++];
    bl[i] = (const float*)d_in[pi++];
    Wr[i] = (const float*)d_in[pi++];
  }
  const float *bg[3], *bb[3];
  for (int i = 0; i < 3; ++i) {
    bg[i] = (const float*)d_in[pi++];
    bb[i] = (const float*)d_in[pi++];
  }
  const float *pw[3], *pb[3];
  for (int i = 0; i < 3; ++i) {
    pw[i] = (const float*)d_in[pi++];
    pb[i] = (const float*)d_in[pi++];
  }
  const float* W1 = (const float*)d_in[pi++];
  const float* b1 = (const float*)d_in[pi++];
  const float* W2 = (const float*)d_in[pi++];
  const float* b2 = (const float*)d_in[pi++];
  const float* W3 = (const float*)d_in[pi++];
  const float* b3 = (const float*)d_in[pi++];

  // --- workspace bump allocator ---
  char* w = (char*)d_ws;
  auto alloc = [&](size_t bytes) {
    char* p = w;
    w += align256(bytes);
    return (void*)p;
  };
  float* msg = (float*)alloc((size_t)NMAX * HDIM * 4);
  float* hbuf = (float*)alloc((size_t)NMAX * HDIM * 4);
  float* xcur = (float*)alloc((size_t)NMAX * HDIM * 4);
  half_t* m16 = (half_t*)alloc((size_t)NMAX * HDIM * 2);
  half_t* x16 = (half_t*)alloc((size_t)NMAX * HDIM * 2);
  half_t* wlT = (half_t*)alloc(HDIM * HDIM * 2);
  half_t* wrT = (half_t*)alloc(HDIM * HDIM * 2);
  float* cnt = (float*)alloc(NMAX * 4);
  float* xw = (float*)alloc(NMAX * 4);
  float* agg = (float*)alloc(NMAX * 4);
  float* score = (float*)alloc(NMAX * 4);
  int* newid = (int*)alloc(NMAX * 4);
  int* perm = (int*)alloc(NMAX * 4);
  int* csrc = (int*)alloc((size_t)NEDGES * 4);
  int* cdst = (int*)alloc((size_t)NEDGES * 4);
  float* bnsum = (float*)alloc(128 * 4);
  float* bnss = (float*)alloc(128 * 4);
  float* xs = (float*)alloc(BGRAPHS * 256 * 4);

  const int per[3] = {500, 400, 320};
  const int kk[3] = {400, 320, 256};

  // init edges + xs accumulator
  edge_init_kernel<<<(NEDGES + 255) / 256, 256, 0, stream>>>(d_ei, csrc, cdst,
                                                             NEDGES);
  zero_f32_kernel<<<256, 256, 0, stream>>>(xs, BGRAPHS * 256);

  for (int L = 0; L < 3; ++L) {
    const int n = BGRAPHS * per[L];
    const int nnew = BGRAPHS * kk[L];
    const float* xin = (L == 0) ? d_x : xcur;
    const float rs = rsqrtf((float)per[L]);

    // zero accumulators
    zero_f32_kernel<<<2048, 256, 0, stream>>>(msg, n * HDIM);
    zero_f32_kernel<<<256, 256, 0, stream>>>(cnt, n);
    zero_f32_kernel<<<256, 256, 0, stream>>>(agg, n);
    zero_f32_kernel<<<1, 256, 0, stream>>>(bnsum, 128);
    zero_f32_kernel<<<1, 256, 0, stream>>>(bnss, 128);

    // mean aggregation (wave per edge, atomic scatter into L2)
    edge_msg_kernel<<<((size_t)NEDGES * 32 + 255) / 256, 256, 0, stream>>>(
        csrc, cdst, xin, msg, cnt, NEDGES);
    mean_fin_kernel<<<(n * HDIM + 255) / 256, 256, 0, stream>>>(msg, cnt, m16,
                                                                n * HDIM);
    cvt_f16_kernel<<<(n * HDIM + 255) / 256, 256, 0, stream>>>(xin, x16,
                                                               n * HDIM);
    weightT_kernel<<<64, 256, 0, stream>>>(Wl[L], wlT);
    weightT_kernel<<<64, 256, 0, stream>>>(Wr[L], wrT);

    // h = relu(mean@Wl + bl + x@Wr) / sqrt(per)   via WMMA
    sage_gemm_kernel<<<n / 16, 256, 0, stream>>>(m16, x16, wlT, wrT, bl[L], rs,
                                                 hbuf);

    // BatchNorm (training-mode batch stats, biased var)
    bn_stats_kernel<<<256, 128, 0, stream>>>(hbuf, bnsum, bnss, n);
    bn_apply_kernel<<<(n * HDIM + 255) / 256, 256, 0, stream>>>(
        hbuf, bnsum, bnss, bg[L], bb[L], 1.0f / (float)n, n * HDIM);

    // SAGPool score: GCNConv(H,1) with self loops + symmetric norm
    xw_kernel<<<((size_t)n * 32 + 255) / 256, 256, 0, stream>>>(hbuf, pw[L],
                                                                xw, n);
    score_edge_kernel<<<(NEDGES + 255) / 256, 256, 0, stream>>>(
        csrc, cdst, xw, cnt, agg, NEDGES);
    score_fin_kernel<<<(n + 255) / 256, 256, 0, stream>>>(agg, xw, cnt, pb[L],
                                                          score, n);

    // per-graph top-k select, gate by tanh(score), relabel edges
    topk_kernel<<<BGRAPHS, 256, 0, stream>>>(score, newid, perm, per[L],
                                             kk[L]);
    gather_kernel<<<((size_t)nnew * 32 + 255) / 256, 256, 0, stream>>>(
        hbuf, score, perm, xcur, nnew);
    relabel_kernel<<<(NEDGES + 255) / 256, 256, 0, stream>>>(csrc, cdst, newid,
                                                             NEDGES);

    // xs += [mean_pool, max_pool]
    pool_kernel<<<BGRAPHS, 128, 0, stream>>>(xcur, xs, kk[L]);
  }

  // final MLP: [200,256] -> 128 -> 64 -> 6
  mlp_kernel<<<BGRAPHS, 256, 0, stream>>>(xs, W1, b1, W2, b2, W3, b3,
                                          (float*)d_out);
}
